// MultiHeadAttention_2164663517185
// MI455X (gfx1250) — matrix operand order repaired and checked by reading, not compiled, not run on hardware
//
#include <hip/hip_runtime.h>
#include <hip/hip_fp16.h>

typedef __attribute__((ext_vector_type(16))) _Float16 v16h;
typedef __attribute__((ext_vector_type(8)))  float    v8f;
typedef __attribute__((ext_vector_type(4)))  _Float16 h4;

#define EMBED 1024
#define NHEAD 16
#define HDIM  64
#define BATCH 4
#define SEQ   2048
#define MROWS (BATCH * SEQ)   // 8192

// ---------------- WMMA fragment helpers (wave32, CDNA5 16x16x32 f16) ----------------
// A-matrix 16(M)x32(K), row-major LDS tile: per-lane data is [8 contiguous] + [8 contiguous]
__device__ __forceinline__ v16h load_frag_a(const _Float16* s, int ldm) {
  const int lane = threadIdx.x & 31;
  const _Float16* p = s + (lane & 15) * ldm + ((lane >> 4) << 4);
  v16h a;
#pragma unroll
  for (int j = 0; j < 8; ++j) {          // k = 0..7 (+16*half)
    a[j] = p[j];
  }
#pragma unroll
  for (int j = 0; j < 8; ++j) {          // k = 8..15 (+16*half)
    a[8 + j] = p[8 + j];
  }
  return a;
}

// B-matrix 32(K)x16(N) from a COLUMN-MAJOR tile: element (k,n) at s[n*ldn + k].
// Per-lane data (n = lane&15, k = 16*half + 0..15) is 16 contiguous halves -> 2x ds_load_b128.
__device__ __forceinline__ v16h load_frag_bt(const _Float16* s, int ldn) {
  const int lane = threadIdx.x & 31;
  const _Float16* p = s + (lane & 15) * ldn + ((lane >> 4) << 4);
  v16h b;
#pragma unroll
  for (int i = 0; i < 16; ++i) b[i] = p[i];
  return b;
}

__device__ __forceinline__ v8f wmma_f16(v16h a, v16h b, v8f c) {
  return __builtin_amdgcn_wmma_f32_16x16x32_f16(false, a, false, b, (short)0, c, false, false);
}

// ---------------- Kernel 0: weight transpose+convert  W[k][n] f32 -> Wh[n][k] f16 ----------
__global__ __launch_bounds__(256) void wtrans_kernel(const float* __restrict__ W,
                                                     _Float16* __restrict__ Wh) {
  int e  = blockIdx.x * 256 + threadIdx.x;   // 1024*256 threads total
  int k  = e >> 8;
  int n4 = (e & 255) * 4;
  float4 v = *(const float4*)&W[(size_t)k * EMBED + n4];
  Wh[(size_t)(n4 + 0) * EMBED + k] = (_Float16)v.x;
  Wh[(size_t)(n4 + 1) * EMBED + k] = (_Float16)v.y;
  Wh[(size_t)(n4 + 2) * EMBED + k] = (_Float16)v.z;
  Wh[(size_t)(n4 + 3) * EMBED + k] = (_Float16)v.w;
}

// ---------------- Kernel 1: QKV projection  Y = X@W + b -> f16 (strided layout) -----------
// block = 256 thr = 8 waves; tile M=128 x N=128; per wave 32(M)x64(N); K-step 32.
__global__ __launch_bounds__(256) void qkv_proj_kernel(
    const float* __restrict__ X,        // [8192,1024] row-major f32
    const _Float16* __restrict__ Wh,    // [N=1024][K=1024] column-major f16
    const float* __restrict__ bias,     // [1024]
    _Float16* __restrict__ dst,
    float scale, int t_stride, int d_stride) {
  __shared__ _Float16 Xs[128][40];      // 128x32 A tile (+pad)
  __shared__ _Float16 WsT[128][40];     // 128 cols x 32 k   (+pad), column-major

  const int tid = threadIdx.x;
  const int wid = tid >> 5;
  const int wm  = wid & 3;              // M quarter (32 rows)
  const int wn  = wid >> 2;             // N half (64 cols)
  const int M0  = blockIdx.y * 128;
  const int N0  = blockIdx.x * 128;

  v8f acc[2][4] = {};

  for (int k0 = 0; k0 < EMBED; k0 += 32) {
    __syncthreads();
    // X tile: 128x32 f32 = 1024 float4; 4/thread; convert + packed h4 store
#pragma unroll
    for (int i = 0; i < 4; ++i) {
      int e = tid + i * 256;
      int r = e >> 3, g = e & 7;
      float4 v = *(const float4*)&X[(size_t)(M0 + r) * EMBED + k0 + g * 4];
      h4 hv = {(_Float16)v.x, (_Float16)v.y, (_Float16)v.z, (_Float16)v.w};
      *(h4*)&Xs[r][g * 4] = hv;
    }
    // W tile: 128 cols x 32 k f16 = 512 uint4; 2/thread (already column-major in Wh)
#pragma unroll
    for (int i = 0; i < 2; ++i) {
      int e = tid + i * 256;
      int c = e >> 2, g = e & 3;
      *(uint4*)&WsT[c][g * 8] = *(const uint4*)&Wh[(size_t)(N0 + c) * EMBED + k0 + g * 8];
    }
    __syncthreads();
    v16h a0 = load_frag_a(&Xs[wm * 32][0], 40);
    v16h a1 = load_frag_a(&Xs[wm * 32 + 16][0], 40);
#pragma unroll
    for (int nt = 0; nt < 4; ++nt) {
      v16h b = load_frag_bt(&WsT[wn * 64 + nt * 16][0], 40);
      acc[0][nt] = wmma_f16(a0, b, acc[0][nt]);
      acc[1][nt] = wmma_f16(a1, b, acc[1][nt]);
    }
  }

  const int lane = tid & 31, half = lane >> 4, nn = lane & 15;
#pragma unroll
  for (int mi = 0; mi < 2; ++mi) {
#pragma unroll
    for (int nt = 0; nt < 4; ++nt) {
#pragma unroll
      for (int r = 0; r < 8; ++r) {
        int m = M0 + wm * 32 + mi * 16 + r + 8 * half;   // [0,8192)
        int n = N0 + wn * 64 + nt * 16 + nn;             // [0,1024)
        float v = (acc[mi][nt][r] + bias[n]) * scale;
        int bb = m >> 11, t = m & (SEQ - 1);
        int hh = n >> 6,  d = n & (HDIM - 1);
        dst[(size_t)(bb * NHEAD + hh) * SEQ * HDIM + (size_t)t * t_stride + (size_t)d * d_stride] =
            (_Float16)v;
      }
    }
  }
}

// ---------------- Kernel 2: causal flash attention ----------------
// block = 128 thr = 4 waves; 64 query rows per block; key chunks of 32. V is [B,H,D,T].
__global__ __launch_bounds__(128) void attn_kernel(
    const _Float16* __restrict__ Qp,  // [B,H,T,D], pre-scaled by 1/sqrt(D)
    const _Float16* __restrict__ Kp,  // [B,H,T,D]
    const _Float16* __restrict__ Vt,  // [B,H,D,T]  (d-major)
    _Float16* __restrict__ Op) {      // [B,T,H*D]
  __shared__ _Float16 Qs[64][72];
  __shared__ _Float16 Ks[32][72];     // [key][d]
  __shared__ _Float16 VsT[64][40];    // [d][key]  (column-major for B operand)
  __shared__ _Float16 Ps[4][16][32];  // per-wave P scratch

  const int tid  = threadIdx.x;
  const int w    = tid >> 5;
  const int lane = tid & 31, half = lane >> 4, nn = lane & 15;
  const int q0 = blockIdx.x * 64;
  const int h  = blockIdx.y;
  const int b  = blockIdx.z;
  const size_t head_off = ((size_t)(b * NHEAD + h)) * SEQ * HDIM;
  const int qbase = q0 + w * 16;

  // stage Q tile 64x64 (512 uint4; 4/thread)
  for (int e = tid; e < 512; e += 128) {
    int r = e >> 3, g = e & 7;
    uint4 v = ((const uint4*)(Qp + head_off + (size_t)(q0 + r) * HDIM))[g];
    *(uint4*)&Qs[r][g * 8] = v;
  }
  __syncthreads();
  v16h qa0 = load_frag_a(&Qs[w * 16][0], 72);
  v16h qa1 = load_frag_a(&Qs[w * 16][32], 72);

  v8f o[4] = {};
  float mrow[8], lrow[8];
#pragma unroll
  for (int r = 0; r < 8; ++r) { mrow[r] = -1.0e30f; lrow[r] = 0.0f; }

  const int nchunks = q0 / 32 + 2;
  for (int ch = 0; ch < nchunks; ++ch) {
    const int kbase = ch * 32;
    __syncthreads();
    // K chunk: 32 keys x 64 d = 256 uint4; 2/thread
    for (int e = tid; e < 256; e += 128) {
      int r = e >> 3, g = e & 7;
      *(uint4*)&Ks[r][g * 8] = ((const uint4*)(Kp + head_off + (size_t)(kbase + r) * HDIM))[g];
    }
    // V chunk (transposed layout): 64 d x 32 keys = 256 uint4; 2/thread
    for (int e = tid; e < 256; e += 128) {
      int d = e >> 2, g = e & 3;
      *(uint4*)&VsT[d][g * 8] =
          *(const uint4*)(Vt + head_off + (size_t)d * SEQ + kbase + g * 8);
    }
    // prefetch next chunk (global_prefetch_b8)
    if (ch + 1 < nchunks) {
      const int nb = kbase + 32;
      __builtin_prefetch(Kp + head_off + (size_t)(nb + (tid >> 2)) * HDIM + (tid & 3) * 16, 0, 3);
      __builtin_prefetch(Vt + head_off + (size_t)(tid >> 1) * SEQ + nb + (tid & 1) * 16, 0, 3);
    }
    __syncthreads();
    if (kbase > qbase + 15) continue;   // fully masked for this wave (wave-uniform)

    // S = Q @ K^T : B operand from Ks (n=key, k=d contiguous)
    v8f s[2];
#pragma unroll
    for (int sub = 0; sub < 2; ++sub) {
      v8f a = {};
      a = wmma_f16(qa0, load_frag_bt(&Ks[sub * 16][0], 72), a);
      a = wmma_f16(qa1, load_frag_bt(&Ks[sub * 16][32], 72), a);
      s[sub] = a;
    }
    // causal mask
#pragma unroll
    for (int r = 0; r < 8; ++r) {
      int row = qbase + r + 8 * half;
#pragma unroll
      for (int sub = 0; sub < 2; ++sub) {
        int col = kbase + sub * 16 + nn;
        if (col > row) s[sub][r] = -3.0e38f;
      }
    }
    // online softmax (row spans 16 lanes within each half)
#pragma unroll
    for (int r = 0; r < 8; ++r) {
      float mx = fmaxf(s[0][r], s[1][r]);
#pragma unroll
      for (int off = 8; off > 0; off >>= 1) mx = fmaxf(mx, __shfl_xor(mx, off));
      float mnew = fmaxf(mrow[r], mx);
      float fac  = __expf(mrow[r] - mnew);
      mrow[r] = mnew;
      float p0 = __expf(s[0][r] - mnew);
      float p1 = __expf(s[1][r] - mnew);
      s[0][r] = p0; s[1][r] = p1;
      float rs = p0 + p1;
#pragma unroll
      for (int off = 8; off > 0; off >>= 1) rs += __shfl_xor(rs, off);
      lrow[r] = lrow[r] * fac + rs;
#pragma unroll
      for (int j = 0; j < 4; ++j) o[j][r] *= fac;
    }
    // P: C-layout -> row-major 16x32 in wave-private LDS (LDS is in-order per wave)
#pragma unroll
    for (int r = 0; r < 8; ++r) {
      int ml = r + 8 * half;
      Ps[w][ml][nn]      = (_Float16)s[0][r];
      Ps[w][ml][16 + nn] = (_Float16)s[1][r];
    }
    __builtin_amdgcn_wave_barrier();
    v16h pa = load_frag_a(&Ps[w][0][0], 32);
    // O += P @ V : B operand from VsT (n=d, k=key contiguous)
#pragma unroll
    for (int j = 0; j < 4; ++j) {
      v16h vb = load_frag_bt(&VsT[j * 16][0], 40);
      o[j] = wmma_f16(pa, vb, o[j]);
    }
  }

  // normalize + store [B,T,H*D] f16
#pragma unroll
  for (int r = 0; r < 8; ++r) {
    int row = qbase + r + 8 * half;
    float inv = 1.0f / lrow[r];
#pragma unroll
    for (int j = 0; j < 4; ++j) {
      int d = j * 16 + nn;
      Op[(((size_t)b * SEQ + row) * NHEAD + h) * HDIM + d] = (_Float16)(o[j][r] * inv);
    }
  }
}

// ---------------- Kernel 3: output projection  out = A@Wo + bo (fp32 out) ----------------
__global__ __launch_bounds__(256) void out_proj_kernel(
    const _Float16* __restrict__ A,    // [8192,1024] f16 row-major
    const _Float16* __restrict__ Wh,   // [N][K] column-major f16
    const float* __restrict__ bias,    // [1024]
    float* __restrict__ out) {         // [8192,1024] fp32
  __shared__ _Float16 As[128][40];
  __shared__ _Float16 WsT[128][40];

  const int tid = threadIdx.x;
  const int wid = tid >> 5;
  const int wm  = wid & 3;
  const int wn  = wid >> 2;
  const int M0  = blockIdx.y * 128;
  const int N0  = blockIdx.x * 128;

  v8f acc[2][4] = {};

  for (int k0 = 0; k0 < EMBED; k0 += 32) {
    __syncthreads();
    // A tile: 128x32 f16 = 512 uint4; 2/thread
#pragma unroll
    for (int i = 0; i < 2; ++i) {
      int e = tid + i * 256;
      int r = e >> 2, g = e & 3;
      *(uint4*)&As[r][g * 8] = *(const uint4*)&A[(size_t)(M0 + r) * EMBED + k0 + g * 8];
    }
    // W tile: 512 uint4; 2/thread
#pragma unroll
    for (int i = 0; i < 2; ++i) {
      int e = tid + i * 256;
      int c = e >> 2, g = e & 3;
      *(uint4*)&WsT[c][g * 8] = *(const uint4*)&Wh[(size_t)(N0 + c) * EMBED + k0 + g * 8];
    }
    __syncthreads();
    v16h a0 = load_frag_a(&As[wm * 32][0], 40);
    v16h a1 = load_frag_a(&As[wm * 32 + 16][0], 40);
#pragma unroll
    for (int nt = 0; nt < 4; ++nt) {
      v16h b = load_frag_bt(&WsT[wn * 64 + nt * 16][0], 40);
      acc[0][nt] = wmma_f16(a0, b, acc[0][nt]);
      acc[1][nt] = wmma_f16(a1, b, acc[1][nt]);
    }
  }

  const int lane = tid & 31, half = lane >> 4, nn = lane & 15;
#pragma unroll
  for (int mi = 0; mi < 2; ++mi) {
#pragma unroll
    for (int nt = 0; nt < 4; ++nt) {
#pragma unroll
      for (int r = 0; r < 8; ++r) {
        int m = M0 + wm * 32 + mi * 16 + r + 8 * half;
        int n = N0 + wn * 64 + nt * 16 + nn;
        out[(size_t)m * EMBED + n] = acc[mi][nt][r] + bias[n];
      }
    }
  }
}

// ---------------- launcher ----------------
extern "C" void kernel_launch(void* const* d_in, const int* in_sizes, int n_in,
                              void* d_out, int out_size, void* d_ws, size_t ws_size,
                              hipStream_t stream) {
  const float* x  = (const float*)d_in[0];
  const float* Wq = (const float*)d_in[1];
  const float* bq = (const float*)d_in[2];
  const float* Wk = (const float*)d_in[3];
  const float* bk = (const float*)d_in[4];
  const float* Wv = (const float*)d_in[5];
  const float* bv = (const float*)d_in[6];
  const float* Wo = (const float*)d_in[7];
  const float* bo = (const float*)d_in[8];
  float* out = (float*)d_out;

  // workspace: Q,K [B,H,T,D] + V [B,H,D,T] + attn-out [B,T,C] (f16) + 4 transposed weights
  const size_t act = (size_t)MROWS * EMBED;   // 8M halves each
  const size_t wel = (size_t)EMBED * EMBED;   // 1M halves each
  _Float16* qbuf = (_Float16*)d_ws;
  _Float16* kbuf = qbuf + act;
  _Float16* vbuf = kbuf + act;
  _Float16* obuf = vbuf + act;
  _Float16* whq  = obuf + act;
  _Float16* whk  = whq + wel;
  _Float16* whv  = whk + wel;
  _Float16* who  = whv + wel;

  // one-shot weight transpose/convert (f32 [K][N] -> f16 [N][K])
  wtrans_kernel<<<1024, 256, 0, stream>>>(Wq, whq);
  wtrans_kernel<<<1024, 256, 0, stream>>>(Wk, whk);
  wtrans_kernel<<<1024, 256, 0, stream>>>(Wv, whv);
  wtrans_kernel<<<1024, 256, 0, stream>>>(Wo, who);

  dim3 gproj(EMBED / 128, MROWS / 128);  // (8, 64)
  // Q,K -> [B,H,T,D] (t*64 + d); V -> [B,H,D,T] (t + d*2048)
  qkv_proj_kernel<<<gproj, 256, 0, stream>>>(x, whq, bq, qbuf, 0.125f, HDIM, 1);
  qkv_proj_kernel<<<gproj, 256, 0, stream>>>(x, whk, bk, kbuf, 1.0f, HDIM, 1);
  qkv_proj_kernel<<<gproj, 256, 0, stream>>>(x, whv, bv, vbuf, 1.0f, 1, SEQ);

  dim3 gattn(SEQ / 64, NHEAD, BATCH);    // (32, 16, 4)
  attn_kernel<<<gattn, 128, 0, stream>>>(qbuf, kbuf, vbuf, obuf);

  out_proj_kernel<<<gproj, 256, 0, stream>>>(obuf, who, bo, out);
}
